// DAWNBlock_21157008900537
// MI455X (gfx1250) — matrix operand order repaired and checked by reading, not compiled, not run on hardware
//
#include <hip/hip_runtime.h>
#include <stdint.h>

#define D_MODEL 1024
#define N_HEADS 16
#define D_HEAD 64
#define RANK 256
#define N_KNOW 4096
#define KC 16
#define KE 8
#define BB 8
#define SS 1024

#if defined(__HIP_DEVICE_COMPILE__) && \
    __has_builtin(__builtin_amdgcn_global_load_async_to_lds_b128) && \
    __has_builtin(__builtin_amdgcn_s_wait_asynccnt)
#define HAVE_ASYNC_LDS 1
#else
#define HAVE_ASYNC_LDS 0
#endif

typedef __attribute__((ext_vector_type(16))) __bf16 v16bf;
typedef __attribute__((ext_vector_type(8)))  float  v8f;

union Frag { v16bf v; unsigned short u[16]; uint4 q[2]; };

__device__ __forceinline__ unsigned short f2bf(float f) {
  unsigned int u = __float_as_uint(f);
  u += 0x7FFFu + ((u >> 16) & 1u);
  return (unsigned short)(u >> 16);
}

#if HAVE_ASYNC_LDS
// Declare the LLVM intrinsic directly (bypasses builtin pointer-AS sema).
// Signature: (ptr addrspace(1) src, ptr addrspace(3) dst, i32 imm offset, i32 cpol)
typedef int v4ig __attribute__((vector_size(16)));
typedef const __attribute__((address_space(1))) v4ig* gptr_t;
typedef __attribute__((address_space(3))) v4ig* lptr_t;
extern "C" __device__ void
amdgcn_async_load_b128(gptr_t gsrc, lptr_t ldst, int offset, int cpol)
    __asm("llvm.amdgcn.global.load.async.to.lds.b128");
// AMDGPU: a flat global address equals the AS1 address; the low 32 bits of a
// flat LDS address are the LDS byte offset (ISA 10.2 aperture calculation).
#define AS1(p) ((gptr_t)(uintptr_t)(p))
#define AS3(p) ((lptr_t)(unsigned int)(uintptr_t)(p))
#endif

// ---------------------------------------------------------------------------
// Generic NT GEMM: C[m,n] = sum_k A[m,k] * B[n,k]; A,B bf16, accum f32.
// One wave computes a 32x32 C macro-tile (2x2 register-blocked 16x16 WMMA
// tiles): 4 v_wmma_f32_16x16x32_bf16 per K-step from 2 A + 2 B fragments.
// When available, A/B panels are staged through LDS with double-buffered
// GLOBAL_LOAD_ASYNC_TO_LDS_B128 (ASYNCcnt) so panel fill overlaps WMMA issue.
// ---------------------------------------------------------------------------
__global__ __launch_bounds__(256) void gemm_nt_kernel(
    const unsigned short* __restrict__ A, const unsigned short* __restrict__ B,
    void* __restrict__ Cv, int M, int N, int K, int lda, int ldb, int ldc,
    long long sA, long long sB, long long sC, int c_bf16)
{
#if HAVE_ASYNC_LDS
  // per wave: 2 phases x (A panel 32x32 halves + B panel 32x32 halves) = 8KB
  __shared__ __align__(16) unsigned short stage[8 * 2 * 2048];
#endif
  int wave = blockIdx.x * (blockDim.x >> 5) + (threadIdx.x >> 5);
  int tX = N >> 5, tY = M >> 5;            // 32x32 macro-tiles
  if (wave >= tX * tY) return;
  int ty = wave / tX, tx = wave % tX;
  int batch = blockIdx.y;
  A += (long long)batch * sA;
  B += (long long)batch * sB;
  int lane = threadIdx.x & 31;
  int hl = lane >> 4;        // half of wave
  int lm = lane & 15;
  int row0 = ty << 5, col0 = tx << 5;
  v8f acc00 = {}, acc01 = {}, acc10 = {}, acc11 = {};

#if HAVE_ASYNC_LDS
  unsigned short* wbase = stage + (threadIdx.x >> 5) * 4096;
  auto issue = [&](int kk, int ph) {
    unsigned short* la = wbase + ph * 2048;    // A: 32 rows x 32 halves
    unsigned short* lb = la + 1024;            // B: 32 cols x 32 halves
    const unsigned short* ga = A + (size_t)(row0 + lane) * lda + kk;
    const unsigned short* gb = B + (size_t)(col0 + lane) * ldb + kk;
    for (int t = 0; t < 4; t++) {
      amdgcn_async_load_b128(AS1(ga + t * 8), AS3(la + lane * 32 + t * 8), 0, 0);
      amdgcn_async_load_b128(AS1(gb + t * 8), AS3(lb + lane * 32 + t * 8), 0, 0);
    }
  };
  issue(0, 0);
  for (int k0 = 0; k0 < K; k0 += 32) {
    int p = (k0 >> 5) & 1;
    if (k0 + 32 < K) {
      issue(k0 + 32, p ^ 1);                   // prefill next phase
      __builtin_amdgcn_s_wait_asynccnt(8);     // current phase complete
    } else {
      __builtin_amdgcn_s_wait_asynccnt(0);
    }
    const unsigned short* la = wbase + p * 2048;
    const unsigned short* lb = la + 1024;
    Frag fa0, fa1, fb0, fb1;
    const unsigned short* qa0p = la + lm * 32 + hl * 16;
    const unsigned short* qa1p = la + (16 + lm) * 32 + hl * 16;
    fa0.q[0] = *(const uint4*)(qa0p); fa0.q[1] = *(const uint4*)(qa0p + 8);
    fa1.q[0] = *(const uint4*)(qa1p); fa1.q[1] = *(const uint4*)(qa1p + 8);
    const unsigned short* qb0p = lb + lm * 32 + hl * 16;
    const unsigned short* qb1p = lb + (16 + lm) * 32 + hl * 16;
    fb0.q[0] = *(const uint4*)(qb0p); fb0.q[1] = *(const uint4*)(qb0p + 8);
    fb1.q[0] = *(const uint4*)(qb1p); fb1.q[1] = *(const uint4*)(qb1p + 8);
    acc00 = __builtin_amdgcn_wmma_f32_16x16x32_bf16(false, fa0.v, false, fb0.v, (short)0, acc00, false, false);
    acc01 = __builtin_amdgcn_wmma_f32_16x16x32_bf16(false, fa0.v, false, fb1.v, (short)0, acc01, false, false);
    acc10 = __builtin_amdgcn_wmma_f32_16x16x32_bf16(false, fa1.v, false, fb0.v, (short)0, acc10, false, false);
    acc11 = __builtin_amdgcn_wmma_f32_16x16x32_bf16(false, fa1.v, false, fb1.v, (short)0, acc11, false, false);
  }
#else
  for (int k0 = 0; k0 < K; k0 += 32) {
    Frag fa0, fa1, fb0, fb1;
    const unsigned short* pa0 = A + (size_t)(row0 + lm) * lda + k0 + hl * 16;
    const unsigned short* pa1 = pa0 + (size_t)16 * lda;
    fa0.q[0] = *(const uint4*)(pa0);
    fa0.q[1] = *(const uint4*)(pa0 + 8);
    fa1.q[0] = *(const uint4*)(pa1);
    fa1.q[1] = *(const uint4*)(pa1 + 8);
    const unsigned short* pb0 = B + (size_t)(col0 + lm) * ldb + k0 + hl * 16;
    const unsigned short* pb1 = pb0 + (size_t)16 * ldb;
    fb0.q[0] = *(const uint4*)(pb0);
    fb0.q[1] = *(const uint4*)(pb0 + 8);
    fb1.q[0] = *(const uint4*)(pb1);
    fb1.q[1] = *(const uint4*)(pb1 + 8);
    if (k0 + 32 < K) {                       // gfx1250 global_prefetch_b8
      __builtin_prefetch(pa0 + 32, 0, 3);
      __builtin_prefetch(pa1 + 32, 0, 3);
      __builtin_prefetch(pb0 + 32, 0, 3);
      __builtin_prefetch(pb1 + 32, 0, 3);
    }
    acc00 = __builtin_amdgcn_wmma_f32_16x16x32_bf16(false, fa0.v, false, fb0.v, (short)0, acc00, false, false);
    acc01 = __builtin_amdgcn_wmma_f32_16x16x32_bf16(false, fa0.v, false, fb1.v, (short)0, acc01, false, false);
    acc10 = __builtin_amdgcn_wmma_f32_16x16x32_bf16(false, fa1.v, false, fb0.v, (short)0, acc10, false, false);
    acc11 = __builtin_amdgcn_wmma_f32_16x16x32_bf16(false, fa1.v, false, fb1.v, (short)0, acc11, false, false);
  }
#endif

  int rb0 = row0 + hl * 8;
  int rb1 = row0 + 16 + hl * 8;
  if (c_bf16) {
    unsigned short* C = (unsigned short*)Cv + (long long)batch * sC;
    for (int i = 0; i < 8; i++) {
      C[(size_t)(rb0 + i) * ldc + col0 + lm]      = f2bf(acc00[i]);
      C[(size_t)(rb0 + i) * ldc + col0 + 16 + lm] = f2bf(acc01[i]);
      C[(size_t)(rb1 + i) * ldc + col0 + lm]      = f2bf(acc10[i]);
      C[(size_t)(rb1 + i) * ldc + col0 + 16 + lm] = f2bf(acc11[i]);
    }
  } else {
    float* C = (float*)Cv + (long long)batch * sC;
    for (int i = 0; i < 8; i++) {
      C[(size_t)(rb0 + i) * ldc + col0 + lm]      = acc00[i];
      C[(size_t)(rb0 + i) * ldc + col0 + 16 + lm] = acc01[i];
      C[(size_t)(rb1 + i) * ldc + col0 + lm]      = acc10[i];
      C[(size_t)(rb1 + i) * ldc + col0 + 16 + lm] = acc11[i];
    }
  }
}

// ---------------------------------------------------------------------------
// f32 -> bf16 conversion
// ---------------------------------------------------------------------------
__global__ void cvt_bf16_kernel(const float* __restrict__ src,
                                unsigned short* __restrict__ dst, int n)
{
  int i = blockIdx.x * 256 + threadIdx.x;
  if (i < n) dst[i] = f2bf(src[i]);
}

// ---------------------------------------------------------------------------
// Router post-processing: softmax over 64 experts per token, importance-
// weighted sum -> w[b,64], renorm, top-k, renorm.  One block per (router,b).
// ---------------------------------------------------------------------------
__global__ __launch_bounds__(256) void router_reduce_kernel(
    const float* __restrict__ logits, const float* __restrict__ importance,
    float* __restrict__ tw, int* __restrict__ ti)
{
  __shared__ float w[64];
  int r = blockIdx.x, b = blockIdx.y;
  int tid = threadIdx.x;
  if (tid < 64) w[tid] = 0.f;
  __syncthreads();
  for (int s = tid; s < SS; s += 256) {
    const float* row = logits + ((size_t)b * SS + s) * 320 + r * 64;
    float mx = -1e30f;
    for (int n = 0; n < 64; n++) mx = fmaxf(mx, row[n]);
    float sum = 0.f;
    for (int n = 0; n < 64; n++) sum += __expf(row[n] - mx);
    float inv = importance[b * SS + s] / sum;
    for (int n = 0; n < 64; n++) atomicAdd(&w[n], __expf(row[n] - mx) * inv);
  }
  __syncthreads();
  if (tid == 0) {
    float tot = 1e-8f;
    for (int n = 0; n < 64; n++) tot += w[n];
    float wl[64];
    for (int n = 0; n < 64; n++) wl[n] = w[n] / tot;
    int k = (r == 0 || r == 4) ? KC : KE;
    float vals[16]; int idx[16];
    float tsum = 1e-8f;
    for (int j = 0; j < k; j++) {
      int best = 0; float bv = -1e30f;
      for (int n = 0; n < 64; n++) if (wl[n] > bv) { bv = wl[n]; best = n; }
      vals[j] = bv; idx[j] = best; wl[best] = -1e30f; tsum += bv;
    }
    int base = (r * BB + b) * 16;
    for (int j = 0; j < k; j++) { tw[base + j] = vals[j] / tsum; ti[base + j] = idx[j]; }
    for (int j = k; j < 16; j++) { tw[base + j] = 0.f; ti[base + j] = 0; }
  }
}

// ---------------------------------------------------------------------------
// shared_c^T[b][rr][d] = sum_k tw[k] * neurons[ti[k]][d][rr]   (bf16 out, NT-ready)
// ---------------------------------------------------------------------------
__global__ void gather_compress_kernel(
    const float* __restrict__ neurons, const float* __restrict__ tw,
    const int* __restrict__ ti, unsigned short* __restrict__ outT, int r)
{
  int b = blockIdx.y;
  int idx = blockIdx.x * 256 + threadIdx.x;   // 0 .. RANK*D_MODEL-1
  int rr = idx >> 10, d = idx & 1023;
  const float* twp = tw + (r * BB + b) * 16;
  const int*   tip = ti + (r * BB + b) * 16;
  float acc = 0.f;
  for (int k = 0; k < KC; k++)
    acc += twp[k] * neurons[((size_t)tip[k] * D_MODEL + d) * RANK + rr];
  outT[(size_t)b * RANK * D_MODEL + (size_t)rr * D_MODEL + d] = f2bf(acc);
}

// ---------------------------------------------------------------------------
// shared_e^T[b][d][rr] = sum_k tw[k] * pool[ti[k]][rr][d]      (bf16 out, NT-ready)
// ---------------------------------------------------------------------------
__global__ void gather_expand_kernel(
    const float* __restrict__ pool, const float* __restrict__ tw,
    const int* __restrict__ ti, unsigned short* __restrict__ outT, int r)
{
  int b = blockIdx.y;
  int idx = blockIdx.x * 256 + threadIdx.x;   // 0 .. D_MODEL*RANK-1
  int d = idx >> 8, rr = idx & 255;
  const float* twp = tw + (r * BB + b) * 16;
  const int*   tip = ti + (r * BB + b) * 16;
  float acc = 0.f;
  for (int k = 0; k < KE; k++)
    acc += twp[k] * pool[((size_t)tip[k] * RANK + rr) * D_MODEL + d];
  outT[(size_t)b * D_MODEL * RANK + (size_t)d * RANK + rr] = f2bf(acc);
}

// ---------------------------------------------------------------------------
// V transpose: Vt[b][d][s] = V[b][s][d]   (bf16)
// ---------------------------------------------------------------------------
__global__ void transpose_v_kernel(const unsigned short* __restrict__ src,
                                   unsigned short* __restrict__ dst)
{
  int b = blockIdx.z;
  int idx = blockIdx.x * 256 + threadIdx.x;   // over D_MODEL*SS
  int d = idx / SS, s = idx % SS;
  dst[(size_t)b * D_MODEL * SS + idx] =
      src[(size_t)b * SS * D_MODEL + (size_t)s * D_MODEL + d];
}

// ---------------------------------------------------------------------------
// Causal flash attention: one wave handles a 16-row Q tile of one (b,head).
// 32-key tiles; S=Q·K^T via 4 WMMAs, online softmax, O+=P·V via 4 WMMAs.
// ---------------------------------------------------------------------------
__global__ __launch_bounds__(32) void attn_kernel(
    const unsigned short* __restrict__ Q, const unsigned short* __restrict__ Kd,
    const unsigned short* __restrict__ Vt, unsigned short* __restrict__ Opre)
{
  __shared__ __align__(16) unsigned short pbuf[16 * 32];
  int b = blockIdx.z, hh = blockIdx.y, qt = blockIdx.x;
  int q0 = qt << 4;
  int hc = hh * D_HEAD;
  int lane = threadIdx.x;
  int hl = lane >> 4, lm = lane & 15;
  int rbase = hl * 8;
  const unsigned short* Qb = Q  + (size_t)b * SS * D_MODEL;
  const unsigned short* Kb = Kd + (size_t)b * SS * D_MODEL;
  const unsigned short* Vb = Vt + (size_t)b * D_MODEL * SS;

  Frag qa0, qa1;                      // Q A-fragments, K-dim 0..31 / 32..63
  {
    const unsigned short* p = Qb + (size_t)(q0 + lm) * D_MODEL + hc + hl * 16;
    qa0.q[0] = *(const uint4*)(p);      qa0.q[1] = *(const uint4*)(p + 8);
    qa1.q[0] = *(const uint4*)(p + 32); qa1.q[1] = *(const uint4*)(p + 40);
  }

  float mrow[8], lrow[8];
  v8f o0 = {}, o1 = {}, o2 = {}, o3 = {};
  for (int i = 0; i < 8; i++) { mrow[i] = -1e30f; lrow[i] = 0.f; }

  int jmax = (q0 + 15) >> 5;
  for (int j = 0; j <= jmax; j++) {
    int kk0 = j << 5;
    v8f st0 = {}, st1 = {};
    {
      const unsigned short* p0 = Kb + (size_t)(kk0 + lm) * D_MODEL + hc + hl * 16;
      const unsigned short* p1 = p0 + (size_t)16 * D_MODEL;
      Frag kb;
      kb.q[0] = *(const uint4*)(p0);      kb.q[1] = *(const uint4*)(p0 + 8);
      st0 = __builtin_amdgcn_wmma_f32_16x16x32_bf16(false, qa0.v, false, kb.v, (short)0, st0, false, false);
      kb.q[0] = *(const uint4*)(p0 + 32); kb.q[1] = *(const uint4*)(p0 + 40);
      st0 = __builtin_amdgcn_wmma_f32_16x16x32_bf16(false, qa1.v, false, kb.v, (short)0, st0, false, false);
      kb.q[0] = *(const uint4*)(p1);      kb.q[1] = *(const uint4*)(p1 + 8);
      st1 = __builtin_amdgcn_wmma_f32_16x16x32_bf16(false, qa0.v, false, kb.v, (short)0, st1, false, false);
      kb.q[0] = *(const uint4*)(p1 + 32); kb.q[1] = *(const uint4*)(p1 + 40);
      st1 = __builtin_amdgcn_wmma_f32_16x16x32_bf16(false, qa1.v, false, kb.v, (short)0, st1, false, false);
    }
    // scale, causal mask, row max
    float rm[8];
    for (int i = 0; i < 8; i++) {
      int row = q0 + rbase + i;
      float v0 = st0[i] * 0.125f, v1 = st1[i] * 0.125f;
      if (kk0 + lm > row)      v0 = -1e30f;
      if (kk0 + 16 + lm > row) v1 = -1e30f;
      st0[i] = v0; st1[i] = v1;
      rm[i] = fmaxf(v0, v1);
    }
    for (int d = 1; d < 16; d <<= 1)
      for (int i = 0; i < 8; i++) rm[i] = fmaxf(rm[i], __shfl_xor(rm[i], d, 32));
    float rs[8];
    for (int i = 0; i < 8; i++) {
      float mn = fmaxf(mrow[i], rm[i]);
      float sc = __expf(mrow[i] - mn);
      float p0e = __expf(st0[i] - mn);
      float p1e = __expf(st1[i] - mn);
      st0[i] = p0e; st1[i] = p1e;
      rs[i] = p0e + p1e;
      mrow[i] = mn;
      lrow[i] = lrow[i] * sc;
      o0[i] = o0[i] * sc; o1[i] = o1[i] * sc;
      o2[i] = o2[i] * sc; o3[i] = o3[i] * sc;
    }
    for (int d = 1; d < 16; d <<= 1)
      for (int i = 0; i < 8; i++) rs[i] += __shfl_xor(rs[i], d, 32);
    for (int i = 0; i < 8; i++) lrow[i] += rs[i];

    // relayout P (C layout -> A layout) through LDS
    __syncthreads();
    for (int i = 0; i < 8; i++) {
      pbuf[(rbase + i) * 32 + lm]      = f2bf(st0[i]);
      pbuf[(rbase + i) * 32 + 16 + lm] = f2bf(st1[i]);
    }
    __syncthreads();
    Frag pa;
    {
      const unsigned short* p = pbuf + lm * 32 + hl * 16;
      pa.q[0] = *(const uint4*)(p);
      pa.q[1] = *(const uint4*)(p + 8);
    }
    for (int t2 = 0; t2 < 4; t2++) {
      Frag vb;
      const unsigned short* p = Vb + (size_t)(hc + t2 * 16 + lm) * SS + kk0 + hl * 16;
      vb.q[0] = *(const uint4*)(p); vb.q[1] = *(const uint4*)(p + 8);
      v8f* op = (t2 == 0) ? &o0 : (t2 == 1) ? &o1 : (t2 == 2) ? &o2 : &o3;
      *op = __builtin_amdgcn_wmma_f32_16x16x32_bf16(false, pa.v, false, vb.v, (short)0, *op, false, false);
    }
    __syncthreads();
  }

  unsigned short* O = Opre + ((size_t)b * SS + q0) * D_MODEL + hc;
  for (int i = 0; i < 8; i++) {
    float inv = 1.f / lrow[i];
    size_t ro = (size_t)(rbase + i) * D_MODEL;
    O[ro +  0 + lm] = f2bf(o0[i] * inv);
    O[ro + 16 + lm] = f2bf(o1[i] * inv);
    O[ro + 32 + lm] = f2bf(o2[i] * inv);
    O[ro + 48 + lm] = f2bf(o3[i] * inv);
  }
}

// ---------------------------------------------------------------------------
// Knowledge lookup: per (b, s) row: top-8 of 4096 scores, softmax(score/16),
// weighted sum of knowledge_V rows, accumulated into out.
// ---------------------------------------------------------------------------
__global__ __launch_bounds__(128) void knowledge_kernel(
    const float* __restrict__ ksc, const float* __restrict__ kV,
    float* __restrict__ out, int b)
{
  __shared__ float cv[128 * 8];
  __shared__ int   ci[128 * 8];
  __shared__ float pw[8];
  __shared__ int   pi[8];
  int s = blockIdx.x, tid = threadIdx.x;
  const float* row = ksc + (size_t)s * N_KNOW;
  float lv[8]; int li[8];
  for (int j = 0; j < 8; j++) { lv[j] = -1e30f; li[j] = 0; }
  for (int n = tid; n < N_KNOW; n += 128) {
    float v = row[n];
    if (v > lv[7]) {
      int j = 7;
      while (j > 0 && v > lv[j - 1]) { lv[j] = lv[j - 1]; li[j] = li[j - 1]; j--; }
      lv[j] = v; li[j] = n;
    }
  }
  for (int j = 0; j < 8; j++) { cv[tid * 8 + j] = lv[j]; ci[tid * 8 + j] = li[j]; }
  __syncthreads();
  if (tid == 0) {
    float gv[8]; int gi[8];
    for (int j = 0; j < 8; j++) { gv[j] = -1e30f; gi[j] = 0; }
    for (int n = 0; n < 128 * 8; n++) {
      float v = cv[n];
      if (v > gv[7]) {
        int j = 7;
        while (j > 0 && v > gv[j - 1]) { gv[j] = gv[j - 1]; gi[j] = gi[j - 1]; j--; }
        gv[j] = v; gi[j] = ci[n];
      }
    }
    float sum = 0.f;
    for (int j = 0; j < 8; j++) {
      float e = __expf((gv[j] - gv[0]) * 0.0625f);   // 1/sqrt(RANK) scale
      pw[j] = e; pi[j] = gi[j]; sum += e;
    }
    for (int j = 0; j < 8; j++) pw[j] /= sum;
  }
  __syncthreads();
  for (int e = tid; e < D_MODEL; e += 128) {
    float acc = 0.f;
    for (int j = 0; j < 8; j++) acc += pw[j] * kV[(size_t)pi[j] * D_MODEL + e];
    out[((size_t)b * SS + s) * D_MODEL + e] += acc;
  }
}

// ---------------------------------------------------------------------------
// Host orchestration
// ---------------------------------------------------------------------------
extern "C" void kernel_launch(void* const* d_in, const int* in_sizes, int n_in,
                              void* d_out, int out_size, void* d_ws, size_t ws_size,
                              hipStream_t stream)
{
  (void)in_sizes; (void)n_in; (void)out_size; (void)ws_size;
  const float* x          = (const float*)d_in[0];
  const float* importance = (const float*)d_in[1];
  const float* compress_w = (const float*)d_in[2];
  const float* expQ_w     = (const float*)d_in[3];
  const float* expK_w     = (const float*)d_in[4];
  const float* expV_w     = (const float*)d_in[5];
  const float* memory_w   = (const float*)d_in[6];
  const float* neurons    = (const float*)d_in[7];
  const float* pool       = (const float*)d_in[8];
  const float* kK         = (const float*)d_in[9];
  const float* kV         = (const float*)d_in[10];
  const float* Wo         = (const float*)d_in[11];
  float* out = (float*)d_out;

  char* ws = (char*)d_ws;
  size_t off = 0;
  auto alloc = [&](size_t bytes) -> void* {
    void* p = (void*)(ws + off);
    off += (bytes + 255) & ~((size_t)255);
    return p;
  };
  unsigned short* Xbf  = (unsigned short*)alloc((size_t)BB * SS * D_MODEL * 2);
  unsigned short* Wall = (unsigned short*)alloc((size_t)5 * 64 * D_MODEL * 2);
  unsigned short* WoBf = (unsigned short*)alloc((size_t)D_MODEL * D_MODEL * 2);
  unsigned short* kKbf = (unsigned short*)alloc((size_t)N_KNOW * RANK * 2);
  float* logits = (float*)alloc((size_t)BB * SS * 320 * 4);
  float* tw     = (float*)alloc((size_t)5 * BB * 16 * 4);
  int*   ti     = (int*)  alloc((size_t)5 * BB * 16 * 4);
  unsigned short* shC  = (unsigned short*)alloc((size_t)BB * RANK * D_MODEL * 2);
  unsigned short* shM  = (unsigned short*)alloc((size_t)BB * RANK * D_MODEL * 2);
  unsigned short* hB   = (unsigned short*)alloc((size_t)BB * SS * RANK * 2);
  unsigned short* hmB  = (unsigned short*)alloc((size_t)BB * SS * RANK * 2);
  unsigned short* shQ  = (unsigned short*)alloc((size_t)BB * D_MODEL * RANK * 2);
  unsigned short* shK  = (unsigned short*)alloc((size_t)BB * D_MODEL * RANK * 2);
  unsigned short* shV  = (unsigned short*)alloc((size_t)BB * D_MODEL * RANK * 2);
  unsigned short* Qbf  = (unsigned short*)alloc((size_t)BB * SS * D_MODEL * 2);
  unsigned short* Kbf  = (unsigned short*)alloc((size_t)BB * SS * D_MODEL * 2);
  unsigned short* Vbf  = (unsigned short*)alloc((size_t)BB * SS * D_MODEL * 2);
  unsigned short* VtB  = (unsigned short*)alloc((size_t)BB * D_MODEL * SS * 2);
  unsigned short* aPre = (unsigned short*)alloc((size_t)BB * SS * D_MODEL * 2);
  float* ksc = (float*)alloc((size_t)SS * N_KNOW * 4);

  auto cvt = [&](const float* s, unsigned short* d, int n) {
    cvt_bf16_kernel<<<(n + 255) / 256, 256, 0, stream>>>(s, d, n);
  };
  auto gemm = [&](const unsigned short* A, const unsigned short* B, void* C,
                  int M, int N, int K, int lda, int ldb, int ldc,
                  long long sA, long long sB, long long sC, int cbf, int batch) {
    int tiles = (M >> 5) * (N >> 5);         // 32x32 macro-tiles, 8 waves/block
    dim3 grid((tiles + 7) / 8, batch);
    gemm_nt_kernel<<<grid, 256, 0, stream>>>(A, B, C, M, N, K, lda, ldb, ldc,
                                             sA, sB, sC, cbf);
  };

  // bf16 conversions
  cvt(x, Xbf, BB * SS * D_MODEL);
  cvt(compress_w, Wall + 0 * 64 * D_MODEL, 64 * D_MODEL);
  cvt(expQ_w,     Wall + 1 * 64 * D_MODEL, 64 * D_MODEL);
  cvt(expK_w,     Wall + 2 * 64 * D_MODEL, 64 * D_MODEL);
  cvt(expV_w,     Wall + 3 * 64 * D_MODEL, 64 * D_MODEL);
  cvt(memory_w,   Wall + 4 * 64 * D_MODEL, 64 * D_MODEL);
  cvt(Wo, WoBf, D_MODEL * D_MODEL);
  cvt(kK, kKbf, N_KNOW * RANK);

  // fused router logits for all 5 routers: [8192, 320]
  gemm(Xbf, Wall, logits, BB * SS, 320, D_MODEL, D_MODEL, D_MODEL, 320, 0, 0, 0, 0, 1);
  router_reduce_kernel<<<dim3(5, BB), 256, 0, stream>>>(logits, importance, tw, ti);

  // compress/memory expert mixes (transposed for NT GEMM)
  gather_compress_kernel<<<dim3(1024, BB), 256, 0, stream>>>(neurons, tw, ti, shC, 0);
  gather_compress_kernel<<<dim3(1024, BB), 256, 0, stream>>>(neurons, tw, ti, shM, 4);

  // h = x @ shared_c, hm = x @ shared_m
  gemm(Xbf, shC, hB, SS, RANK, D_MODEL, D_MODEL, D_MODEL, RANK,
       (long long)SS * D_MODEL, (long long)RANK * D_MODEL, (long long)SS * RANK, 1, BB);
  gemm(Xbf, shM, hmB, SS, RANK, D_MODEL, D_MODEL, D_MODEL, RANK,
       (long long)SS * D_MODEL, (long long)RANK * D_MODEL, (long long)SS * RANK, 1, BB);

  // expand mixes
  gather_expand_kernel<<<dim3(1024, BB), 256, 0, stream>>>(pool, tw, ti, shQ, 1);
  gather_expand_kernel<<<dim3(1024, BB), 256, 0, stream>>>(pool, tw, ti, shK, 2);
  gather_expand_kernel<<<dim3(1024, BB), 256, 0, stream>>>(pool, tw, ti, shV, 3);

  // Q/K/V = h @ shared_{q,k,v}
  gemm(hB, shQ, Qbf, SS, D_MODEL, RANK, RANK, RANK, D_MODEL,
       (long long)SS * RANK, (long long)D_MODEL * RANK, (long long)SS * D_MODEL, 1, BB);
  gemm(hB, shK, Kbf, SS, D_MODEL, RANK, RANK, RANK, D_MODEL,
       (long long)SS * RANK, (long long)D_MODEL * RANK, (long long)SS * D_MODEL, 1, BB);
  gemm(hB, shV, Vbf, SS, D_MODEL, RANK, RANK, RANK, D_MODEL,
       (long long)SS * RANK, (long long)D_MODEL * RANK, (long long)SS * D_MODEL, 1, BB);

  transpose_v_kernel<<<dim3((D_MODEL * SS) / 256, 1, BB), 256, 0, stream>>>(Vbf, VtB);

  // causal flash attention
  attn_kernel<<<dim3(SS / 16, N_HEADS, BB), 32, 0, stream>>>(Qbf, Kbf, VtB, aPre);

  // output projection into d_out
  gemm(aPre, WoBf, out, BB * SS, D_MODEL, D_MODEL, D_MODEL, D_MODEL, D_MODEL,
       0, 0, 0, 0, 1);

  // knowledge path, chunked per batch to bound workspace
  for (int b = 0; b < BB; b++) {
    gemm(hmB + (size_t)b * SS * RANK, kKbf, ksc, SS, N_KNOW, RANK, RANK, RANK,
         N_KNOW, 0, 0, 0, 0, 1);
    knowledge_kernel<<<SS, 128, 0, stream>>>(ksc, kV, out, b);
  }
}
